// SNN_15917148799596
// MI455X (gfx1250) — matrix operand order repaired and checked by reading, not compiled, not run on hardware
//
#include <hip/hip_runtime.h>
#include <hip/hip_bf16.h>

// ---------------------------------------------------------------------------
// CDNA5 / gfx1250 SNN pipeline (wave32, WMMA f16).
//   K1 init        : zero histogram / counters / reduction scalars
//   K2 prepA       : pack W[100,4,3,3] -> f16 A-operand in ISA 16x32 layout
//   K3 gabor       : 4-orientation 3x3 Gabor + abs + channel winner-take-all
//   K4 hist        : 32768-bin intensity histogram (descending-rank prep)
//   K5 scan        : suffix-sum scan, nonzero count, bin_size
//   K6 assign      : per-pixel spike time (0..14, 127 = never)
//   K7 minpool     : 5x5 pad-1 min over spike times -> m[4,252,252] u8
//   K8 gemm (WMMA) : pot[t,f,y,x] = W . spikes(m<=t), threshold, store thr
//   K9/K10 winner  : n_act / first-spike potential, global v, packed argmax
//   K11 finalize   : class decision -> d_out[0]
// ---------------------------------------------------------------------------

typedef __attribute__((ext_vector_type(16))) _Float16 v16h;
typedef __attribute__((ext_vector_type(8)))  float    v8f;

#define T_BINSK 15
#define THRESHK 20.0f
#define H_IMG   256
#define H1      254                 // after 3x3 valid gabor
#define NPIX1   (H1 * H1)           // 64516
#define NFEATK  (4 * NPIX1)         // 258064
#define H2      252                 // after 5x5 pool pad 1
#define NPIX2   (H2 * H2)           // 63504
#define H3      250                 // after 3x3 valid spiking conv
#define NPIX3   (H3 * H3)           // 62500
#define NFK     100
#define NPOSK   (NFK * NPIX3)       // 6,250,000
#define NBINS   32768
#define NTILES  ((NPIX3 + 15) / 16) // 3907
#define MTILES  7                   // ceil(100/16)
#define A_HALVES (2 * MTILES * 32 * 16) // 2 K-steps * 7 tiles * 32 lanes * 16 halves

// workspace byte offsets (all suitably aligned)
#define WS_FEAT   0u          // 258064 f32
#define WS_ST     1032256u    // 258064 u8 spike times
#define WS_M      1290320u    // 254016 u8 min-pooled spike times
#define WS_HIST   1544336u    // 32768 u32
#define WS_CNT    1675408u    // 32768 u32
#define WS_SUF    1806480u    // 32768 u32
#define WS_SCAL   1937552u    // 16 u32: [0]=nonzero [1]=bin_size [2]=vmax bits [4..5]=pack u64
#define WS_A16    1937664u    // 7168 f16 (32B aligned)

// ------------------------------- K1: init ----------------------------------
__global__ void snn_init(unsigned* hist, unsigned* cnt, unsigned* scal) {
  int i = blockIdx.x * blockDim.x + threadIdx.x;
  if (i < NBINS) { hist[i] = 0u; cnt[i] = 0u; }
  if (i < 8) scal[i] = 0u;
}

// ------------------------------- K2: pack A --------------------------------
// ISA 16-bit A-matrix 16x32 layout (CDNA5 ISA):
//   lanes 0-15 : row M=lane,    halves 0-7 -> K=0..7,  halves 8-15 -> K=16..23
//   lanes 16-31: row M=lane-16, halves 0-7 -> K=8..15, halves 8-15 -> K=24..31
// Flatten: A16[((ks*7 + tile)*32 + lane)*16 + h], K_global = ks*32 + K_local.
// im2col tap index K = c*9 + dy*3 + dx matches OIHW W flat layout (row*36 + K).
__global__ void snn_prepA(const float* __restrict__ W, _Float16* __restrict__ A16) {
  int gid = blockIdx.x * blockDim.x + threadIdx.x;
  if (gid >= A_HALVES) return;
  int h    = gid & 15;
  int lane = (gid >> 4) & 31;
  int tile = (gid >> 9) % MTILES;
  int ks   = gid / (MTILES * 512);
  int Kl   = ((h < 8) ? h : h + 8) + ((lane < 16) ? 0 : 8);
  int K    = ks * 32 + Kl;
  int row  = tile * 16 + (lane & 15);
  float v  = 0.f;
  if (row < NFK && K < 36) v = W[row * 36 + K];
  A16[gid] = (_Float16)v;
}

// ------------------------------- K3: gabor ---------------------------------
__global__ void __launch_bounds__(256)
snn_gabor(const float* __restrict__ img, float* __restrict__ feat) {
  __shared__ float G[4][9];
  if (threadIdx.x < 4) {
    const float orient[4] = {67.5f, 112.5f, 157.5f, 202.5f};
    const float th = orient[threadIdx.x] * 0.017453292519943295f;
    const float lam = 1.5f, sig_sq = 1.44f, g = 0.3f;
    float tmp[9]; float mean = 0.f;
    for (int xi = -1; xi <= 1; ++xi)
      for (int yi = -1; yi <= 1; ++yi) {
        float Yr = (float)yi * __cosf(th) - (float)xi * __sinf(th);
        float Xr = (float)yi * __sinf(th) + (float)xi * __cosf(th);
        float val = __expf(-(Xr * Xr + g * g * Yr * Yr) / (2.f * sig_sq)) *
                    __cosf(6.283185307179586f * Xr / lam);
        tmp[(xi + 1) * 3 + (yi + 1)] = val;
        mean += val;
      }
    mean *= (1.f / 9.f);
    float mx = 0.f;
    for (int k = 0; k < 9; ++k) { tmp[k] -= mean; mx = fmaxf(mx, fabsf(tmp[k])); }
    float inv = 1.f / mx;
    for (int k = 0; k < 9; ++k) G[threadIdx.x][k] = tmp[k] * inv;
  }
  __syncthreads();
  int p = blockIdx.x * blockDim.x + threadIdx.x;
  if (p >= NPIX1) return;
  int y = p / H1, x = p % H1;
  float a[4]; int best = 0; float bestv = -1.f;
  #pragma unroll
  for (int o = 0; o < 4; ++o) {
    float s = 0.f;
    #pragma unroll
    for (int i = 0; i < 3; ++i)
      #pragma unroll
      for (int j = 0; j < 3; ++j)
        s += img[(y + i) * H_IMG + (x + j)] * G[o][i * 3 + j];
    a[o] = fabsf(s);
    if (a[o] > bestv) { bestv = a[o]; best = o; }
  }
  #pragma unroll
  for (int o = 0; o < 4; ++o)
    feat[o * NPIX1 + p] = (o == best && bestv > 0.f) ? a[o] : 0.f;
}

// ------------------------------- K4: histogram -----------------------------
__device__ __forceinline__ int quantize_bin(float v) {
  int q = (int)(v * ((float)NBINS / 9.01f));   // |feat| <= sum|G| < 9
  return q >= NBINS ? NBINS - 1 : q;
}

__global__ void snn_hist(const float* __restrict__ feat, unsigned* hist) {
  int i = blockIdx.x * blockDim.x + threadIdx.x;
  if (i >= NFEATK) return;
  float v = feat[i];
  if (v > 0.f) atomicAdd(&hist[quantize_bin(v)], 1u);
}

// ------------------------------- K5: suffix scan ---------------------------
__global__ void __launch_bounds__(1024)
snn_scan(const unsigned* __restrict__ hist, unsigned* __restrict__ suffix,
         unsigned* scal) {
  __shared__ unsigned sc[1024];
  const int t = threadIdx.x;
  unsigned s = 0;
  for (int i = 0; i < 32; ++i) s += hist[t * 32 + i];
  const unsigned mine = s;
  sc[t] = s;
  __syncthreads();
  for (int off = 1; off < 1024; off <<= 1) {   // inclusive suffix scan
    unsigned add = (t + off < 1024) ? sc[t + off] : 0u;
    __syncthreads();
    sc[t] += add;
    __syncthreads();
  }
  unsigned run = sc[t] - mine;                 // count of elements in higher bins
  for (int i = 31; i >= 0; --i) {
    int bin = t * 32 + i;
    suffix[bin] = run;                         // descending rank base
    run += hist[bin];
  }
  if (t == 0) {
    unsigned total = sc[0];
    scal[0] = total;
    unsigned bs = total / T_BINSK;
    scal[1] = bs < 1u ? 1u : bs;
  }
}

// ------------------------------- K6: spike times ---------------------------
__global__ void snn_assign(const float* __restrict__ feat,
                           const unsigned* __restrict__ suffix, unsigned* cnt,
                           const unsigned* __restrict__ scal,
                           unsigned char* __restrict__ st) {
  int i = blockIdx.x * blockDim.x + threadIdx.x;
  if (i >= NFEATK) return;
  unsigned char out = 127;
  float v = feat[i];
  if (v > 0.f) {
    int q = quantize_bin(v);
    unsigned r = suffix[q] + atomicAdd(&cnt[q], 1u);
    unsigned stt = r / scal[1];
    if (stt < T_BINSK) out = (unsigned char)stt;
  }
  st[i] = out;
}

// ------------------------------- K7: min-pool 5x5 --------------------------
// pooled[t] = (min spike time over window) <= t  -> keep only the min map.
__global__ void snn_minpool(const unsigned char* __restrict__ st,
                            unsigned char* __restrict__ m) {
  int i = blockIdx.x * blockDim.x + threadIdx.x;
  if (i >= 4 * NPIX2) return;
  int c = i / NPIX2, rem = i % NPIX2;
  int yo = rem / H2, xo = rem % H2;
  unsigned best = 127u;
  for (int ky = 0; ky < 5; ++ky) {
    int yy = yo - 1 + ky;
    if (yy < 0 || yy >= H1) continue;
    for (int kx = 0; kx < 5; ++kx) {
      int xx = xo - 1 + kx;
      if (xx < 0 || xx >= H1) continue;
      unsigned v = st[c * NPIX1 + yy * H1 + xx];
      best = v < best ? v : best;
    }
  }
  m[i] = (unsigned char)best;
}

// ------------------------------- K8: WMMA GEMM -----------------------------
// One block per (16-pixel N-tile, timestep t). 256 threads build the shared
// 32x16 f16 B operand (2 K-steps) from the u8 min-spike-time map, then waves
// 0..6 each run two V_WMMA_F32_16X16X32_F16 (K=36 padded to 64) against the
// pre-packed A tiles and store thresholded potentials.
// ISA 16-bit B-matrix 32x16 layout: lanes 0-15 col=lane K=0..7,16..23,
// lanes 16-31 col=lane-16 K=8..15,24..31 (same K order as A).
// Epilogue: waves 0-5 (features 0..95) and all but the last N-tile take a
// branch-free store path; only wave 6 / the edge tile keep per-element guards.
__global__ void __launch_bounds__(256)
snn_gemm(const _Float16* __restrict__ A16, const unsigned char* __restrict__ m,
         float* __restrict__ thr) {
  const int ntile = blockIdx.x;
  const int t     = blockIdx.y;
  const int tid   = threadIdx.x;
  __shared__ v16h Bsh[2][32];
  {
    _Float16* bp = (_Float16*)Bsh;
    #pragma unroll
    for (int u = 0; u < 4; ++u) {
      int hf   = tid * 4 + u;        // 0..1023 over [kstep][lane][half]
      int ks   = hf >> 9;
      int lane = (hf >> 4) & 31;
      int h    = hf & 15;
      int Kl   = ((h < 8) ? h : h + 8) + ((lane < 16) ? 0 : 8);
      int K    = ks * 32 + Kl;
      int n    = ntile * 16 + (lane & 15);
      _Float16 val = (_Float16)0.f;
      if (K < 36 && n < NPIX3) {
        int y = n / H3, x = n % H3;
        int c = K / 9, rr = (K % 9) / 3, cc = K % 3;
        if ((int)m[c * NPIX2 + (y + rr) * H2 + (x + cc)] <= t)
          val = (_Float16)1.0f;      // binary spike, exact in f16
      }
      bp[hf] = val;
    }
  }
  __syncthreads();
  const int wave = tid >> 5, lane = tid & 31;
  if (wave < MTILES) {
    v8f c = {0.f, 0.f, 0.f, 0.f, 0.f, 0.f, 0.f, 0.f};
    const v16h* Ah = (const v16h*)A16;
    #pragma unroll
    for (int ks = 0; ks < 2; ++ks) {
      v16h a = Ah[(ks * MTILES + wave) * 32 + lane];
      v16h b = Bsh[ks][lane];
      c = __builtin_amdgcn_wmma_f32_16x16x32_f16(false, a, false, b,
                                                 (short)0, c, false, false);
    }
    // C/D layout: lanes 0-15 -> N=lane, M=vgpr; lanes 16-31 -> N=lane-16, M=8+vgpr
    const int n     = ntile * 16 + (lane & 15);
    const int mhalf = (lane < 16) ? 0 : 8;
    const int fbase = wave * 16 + mhalf;
    if (wave < MTILES - 1 && ntile < NTILES - 1) {
      // uniform fast path: all 8 rows and all 16 columns in range
      float* outp = thr + (size_t)t * NPOSK + (size_t)fbase * NPIX3 + n;
      #pragma unroll
      for (int v = 0; v < 8; ++v) {
        float pot = c[v];
        outp[(size_t)v * NPIX3] = (pot < THRESHK) ? 0.f : pot;
      }
    } else {
      #pragma unroll
      for (int v = 0; v < 8; ++v) {
        const int f = fbase + v;
        if (f < NFK && n < NPIX3) {
          float pot = c[v];
          thr[(size_t)(t * NFK + f) * NPIX3 + n] = (pot < THRESHK) ? 0.f : pot;
        }
      }
    }
  }
}

// ------------------------------- K9/K10: winner ----------------------------
__global__ void __launch_bounds__(256)
snn_winner1(const float* __restrict__ thr, unsigned* scal) {
  const int p = blockIdx.x * blockDim.x + threadIdx.x;
  float cand = 0.f;
  if (p < NPOSK) {
    // warm the deepest plane before the dependent 15-load chain
    __builtin_prefetch(&thr[(size_t)(T_BINSK - 1) * NPOSK + p], 0, 1);
    float v[T_BINSK]; int n_act = 0;
    #pragma unroll
    for (int t = 0; t < T_BINSK; ++t) {
      v[t] = thr[(size_t)t * NPOSK + p];
      if (v[t] > 0.f) ++n_act;
    }
    if (n_act > 0) {
      int tf = T_BINSK - n_act;
      if (tf < 0) tf = 0; if (tf > T_BINSK - 1) tf = T_BINSK - 1;
      cand = v[tf];                  // potential at first spike time
    }
  }
  __shared__ float red[256];
  red[threadIdx.x] = cand;
  __syncthreads();
  for (int s = 128; s > 0; s >>= 1) {
    if (threadIdx.x < s) red[threadIdx.x] = fmaxf(red[threadIdx.x], red[threadIdx.x + s]);
    __syncthreads();
  }
  if (threadIdx.x == 0)
    atomicMax(&scal[2], __float_as_uint(red[0]));   // vals >= 0: bit-order == value-order
}

__global__ void __launch_bounds__(256)
snn_winner2(const float* __restrict__ thr, unsigned* scal) {
  const int p = blockIdx.x * blockDim.x + threadIdx.x;
  unsigned long long pack = 0ull;
  if (p < NPOSK) {
    __builtin_prefetch(&thr[(size_t)(T_BINSK - 1) * NPOSK + p], 0, 1);
    const float vglob = __uint_as_float(scal[2]) * (float)T_BINSK;
    float v[T_BINSK]; int n_act = 0;
    #pragma unroll
    for (int t = 0; t < T_BINSK; ++t) {
      v[t] = thr[(size_t)t * NPOSK + p];
      if (v[t] > 0.f) ++n_act;
    }
    float total = 0.f;
    if (n_act > 0) {
      int tf = T_BINSK - n_act;
      if (tf < 0) tf = 0; if (tf > T_BINSK - 1) tf = T_BINSK - 1;
      total = (float)n_act * v[tf] + (float)n_act * vglob;
    }
    pack = ((unsigned long long)__float_as_uint(total) << 32) |
           (unsigned long long)(0xFFFFFFFFu - (unsigned)p);   // ties -> lowest index
  }
  __shared__ unsigned long long red[256];
  red[threadIdx.x] = pack;
  __syncthreads();
  for (int s = 128; s > 0; s >>= 1) {
    if (threadIdx.x < s && red[threadIdx.x + s] > red[threadIdx.x])
      red[threadIdx.x] = red[threadIdx.x + s];
    __syncthreads();
  }
  if (threadIdx.x == 0)
    atomicMax((unsigned long long*)(scal + 4), red[0]);
}

// ------------------------------- K11: finalize -----------------------------
__global__ void snn_finalize(const unsigned* __restrict__ scal, float* out) {
  unsigned long long pack = *(const unsigned long long*)(scal + 4);
  unsigned hi = (unsigned)(pack >> 32);
  unsigned p  = 0xFFFFFFFFu - (unsigned)(pack & 0xFFFFFFFFull);
  int feature = (int)(p / NPIX3);
  int cls     = feature / 50;                 // FPC = 50
  out[0] = (hi != 0u) ? (float)cls : -1.0f;
}

// ------------------------------- launch ------------------------------------
extern "C" void kernel_launch(void* const* d_in, const int* in_sizes, int n_in,
                              void* d_out, int out_size, void* d_ws, size_t ws_size,
                              hipStream_t stream) {
  (void)in_sizes; (void)n_in; (void)out_size; (void)ws_size;
  const float* x = (const float*)d_in[0];     // [1,256,256]
  const float* W = (const float*)d_in[1];     // [100,4,3,3]
  float* outb = (float*)d_out;                // [0]=class, [1..]=thr
  float* thr  = outb + 1;

  char* w = (char*)d_ws;
  float*         feat   = (float*)(w + WS_FEAT);
  unsigned char* st     = (unsigned char*)(w + WS_ST);
  unsigned char* mmap   = (unsigned char*)(w + WS_M);
  unsigned*      hist   = (unsigned*)(w + WS_HIST);
  unsigned*      cnt    = (unsigned*)(w + WS_CNT);
  unsigned*      suffix = (unsigned*)(w + WS_SUF);
  unsigned*      scal   = (unsigned*)(w + WS_SCAL);
  _Float16*      A16    = (_Float16*)(w + WS_A16);

  snn_init   <<<(NBINS + 255) / 256, 256, 0, stream>>>(hist, cnt, scal);
  snn_prepA  <<<(A_HALVES + 255) / 256, 256, 0, stream>>>(W, A16);
  snn_gabor  <<<(NPIX1 + 255) / 256, 256, 0, stream>>>(x, feat);
  snn_hist   <<<(NFEATK + 255) / 256, 256, 0, stream>>>(feat, hist);
  snn_scan   <<<1, 1024, 0, stream>>>(hist, suffix, scal);
  snn_assign <<<(NFEATK + 255) / 256, 256, 0, stream>>>(feat, suffix, cnt, scal, st);
  snn_minpool<<<(4 * NPIX2 + 255) / 256, 256, 0, stream>>>(st, mmap);
  dim3 ggrid(NTILES, T_BINSK);
  snn_gemm   <<<ggrid, 256, 0, stream>>>(A16, mmap, thr);
  snn_winner1<<<(NPOSK + 255) / 256, 256, 0, stream>>>(thr, scal);
  snn_winner2<<<(NPOSK + 255) / 256, 256, 0, stream>>>(thr, scal);
  snn_finalize<<<1, 1, 0, stream>>>(scal, outb);
}
